// QCGN2oEI_Minimal_6287832122018
// MI455X (gfx1250) — matrix operand order repaired and checked by reading, not compiled, not run on hardware
//
#include <hip/hip_runtime.h>
#include <hip/hip_bf16.h>

// ---------------------------------------------------------------------------
// Problem constants (from reference)
// ---------------------------------------------------------------------------
constexpr int N_NODES  = 10000;
constexpr int N_EDGES  = 40000;
constexpr int N_GRAPHS = 200;
constexpr int ND  = 34;
constexpr int ED  = 10;
constexpr int HID = 768;
constexpr int LAYERS = 14;
constexpr int HEADS  = 24;
constexpr int DHEAD  = 32;           // HID / HEADS
constexpr int OUT = 1000;
constexpr int EA  = N_EDGES + N_NODES;   // edges + self loops = 50000

// ---------------------------------------------------------------------------
// Helpers
// ---------------------------------------------------------------------------
typedef __attribute__((ext_vector_type(16))) __bf16 v16bf;
typedef __attribute__((ext_vector_type(2)))  __bf16 v2bf;
typedef __attribute__((ext_vector_type(8)))  float  v8f;

union BF16Frag { v16bf v; unsigned u[8]; };

// fp32 -> packed 2x bf16 via the __bf16 type: clang lowers fptrunc to the
// native bf16 convert (v_cvt_pk_bf16_f32 on targets that have it), RNE.
__device__ __forceinline__ unsigned pack2bf(float lo, float hi) {
  v2bf p;
  p.x = (__bf16)lo;
  p.y = (__bf16)hi;
  unsigned u;
  __builtin_memcpy(&u, &p, sizeof(u));
  return u;
}

__device__ __forceinline__ float elu1(float v) {
  return v > 0.f ? v : (__expf(v) - 1.f);
}

__device__ __forceinline__ void atomicMaxF(float* addr, float val) {
  if (val >= 0.f) atomicMax((int*)addr, __float_as_int(val));
  else            atomicMin((unsigned int*)addr, __float_as_uint(val));
}

// ---------------------------------------------------------------------------
// BF16 WMMA GEMM:  C[M,N] = act( A[M,K] * B[K,N] + bias[N] )
// K multiple of 32.  128x128 tile per 256-thread block (8 waves, 2x4 wave
// grid).  Each wave: 64 rows x 32 cols = 4x2 WMMA tiles -> 8x
// v_wmma_f32_16x16x32_bf16 per K-step.  fp32 -> bf16 conversion in-flight;
// LDS double-buffered with register staging: one barrier per K-step.
// ---------------------------------------------------------------------------
__global__ void __launch_bounds__(256)
k_gemm_bf16(const float* __restrict__ A, const float* __restrict__ B,
            const float* __restrict__ bias, float* __restrict__ C,
            int M, int K, int N, int act) {
  __shared__ unsigned short As[2][128][32];   // [buf][row][k]  8KB per buf
  __shared__ unsigned short Bs[2][128][32];   // [buf][col][k]  (B transposed)

  const int tid  = threadIdx.x;
  const int wave = tid >> 5;      // 0..7
  const int lane = tid & 31;
  const int half = lane >> 4;
  const int l15  = lane & 15;
  const int wr   = wave >> 2;     // 0..1 : 64-row group
  const int wc   = wave & 3;      // 0..3 : 32-col group
  const int row0 = blockIdx.x * 128;
  const int col0 = blockIdx.y * 128;

  // A staging: thread -> row = tid>>1, 16 floats at k = (tid&1)*16
  const int rowA = tid >> 1;
  const int segA = (tid & 1) * 16;
  // B staging: thread -> 8 cols at (tid&15)*8, 2 ks at (tid>>4)*2
  const int colB = (tid & 15) * 8;
  const int kB   = (tid >> 4) * 2;

  float aR[16];
  float bR[2][8];
  v8f acc[4][2] = {};
  const int T = K / 32;

  auto load_regs = [&](int t) {
    const int k0 = t * 32;
    const int gr = row0 + rowA;
    if (gr < M) {
      const float4* p = (const float4*)(A + (size_t)gr * K + k0 + segA);
#pragma unroll
      for (int i = 0; i < 4; ++i) {
        float4 v = p[i];
        aR[4 * i + 0] = v.x; aR[4 * i + 1] = v.y;
        aR[4 * i + 2] = v.z; aR[4 * i + 3] = v.w;
      }
    } else {
#pragma unroll
      for (int i = 0; i < 16; ++i) aR[i] = 0.f;
    }
#pragma unroll
    for (int r = 0; r < 2; ++r) {
      const int gk = k0 + kB + r;
      const int gc = col0 + colB;
      if (gc + 7 < N) {
        const float4* p = (const float4*)(B + (size_t)gk * N + gc);
        float4 v0 = p[0], v1 = p[1];
        bR[r][0] = v0.x; bR[r][1] = v0.y; bR[r][2] = v0.z; bR[r][3] = v0.w;
        bR[r][4] = v1.x; bR[r][5] = v1.y; bR[r][6] = v1.z; bR[r][7] = v1.w;
      } else {
#pragma unroll
        for (int j = 0; j < 8; ++j) {
          int c = gc + j;
          bR[r][j] = (c < N) ? B[(size_t)gk * N + c] : 0.f;
        }
      }
    }
  };

  auto store_lds = [&](int buf) {
    uint4 w0, w1;
    w0.x = pack2bf(aR[0],  aR[1]);  w0.y = pack2bf(aR[2],  aR[3]);
    w0.z = pack2bf(aR[4],  aR[5]);  w0.w = pack2bf(aR[6],  aR[7]);
    w1.x = pack2bf(aR[8],  aR[9]);  w1.y = pack2bf(aR[10], aR[11]);
    w1.z = pack2bf(aR[12], aR[13]); w1.w = pack2bf(aR[14], aR[15]);
    uint4* pa = (uint4*)&As[buf][rowA][segA];
    pa[0] = w0; pa[1] = w1;
#pragma unroll
    for (int j = 0; j < 8; ++j)
      *(unsigned*)&Bs[buf][colB + j][kB] = pack2bf(bR[0][j], bR[1][j]);
  };

  auto compute = [&](int buf) {
    // B fragments (contiguous k order used here: VGPR i holds K = 2i,2i+1 for lanes 0-15,
    // K = 16+2i,17+2i for lanes 16-31) -> kkB = 2i + 16*half
    BF16Frag bfr[2];
#pragma unroll
    for (int nt = 0; nt < 2; ++nt) {
      const int bcol = wc * 32 + nt * 16 + l15;
#pragma unroll
      for (int i = 0; i < 8; ++i) {
        int kk = (i << 1) + (half << 4);
        bfr[nt].u[i] = *(const unsigned*)&Bs[buf][bcol][kk];
      }
    }
#pragma unroll
    for (int mt = 0; mt < 4; ++mt) {
      // A fragment (same contiguous k order as B): kkA = 2i + 16*half
      BF16Frag a;
      const int arow = wr * 64 + mt * 16 + l15;
#pragma unroll
      for (int i = 0; i < 8; ++i) {
        int kk = (i << 1) + (half << 4);
        a.u[i] = *(const unsigned*)&As[buf][arow][kk];
      }
#pragma unroll
      for (int nt = 0; nt < 2; ++nt) {
        acc[mt][nt] = __builtin_amdgcn_wmma_f32_16x16x32_bf16(
            /*neg_a=*/false, a.v, /*neg_b=*/false, bfr[nt].v,
            /*c_mod=*/(short)0, acc[mt][nt],
            /*reuse_a=*/false, /*reuse_b=*/false);
      }
    }
  };

  load_regs(0);
  store_lds(0);
  __syncthreads();
  for (int t = 0; t < T; ++t) {
    const int buf = t & 1;
    if (t + 1 < T) load_regs(t + 1);
    compute(buf);
    if (t + 1 < T) store_lds(buf ^ 1);
    __syncthreads();
  }

  // store: f32 C/D layout -> VGPR j: row j (lanes 0-15) / j+8 (lanes 16-31)
#pragma unroll
  for (int mt = 0; mt < 4; ++mt) {
#pragma unroll
    for (int nt = 0; nt < 2; ++nt) {
      const int gc = col0 + wc * 32 + nt * 16 + l15;
      if (gc >= N) continue;
      const float bv = bias ? bias[gc] : 0.f;
#pragma unroll
      for (int j = 0; j < 8; ++j) {
        const int gr = row0 + wr * 64 + mt * 16 + j + half * 8;
        if (gr < M) {
          float v = acc[mt][nt][j] + bv;
          if (act) v = elu1(v);
          C[(size_t)gr * N + gc] = v;
        }
      }
    }
  }
}

// ---------------------------------------------------------------------------
// Small-K linear (encoders): one thread per output element
// ---------------------------------------------------------------------------
__global__ void k_linear_small(const float* __restrict__ A,
                               const float* __restrict__ W,
                               const float* __restrict__ b,
                               float* __restrict__ C,
                               int M, int K, int N, int act) {
  long long idx = (long long)blockIdx.x * blockDim.x + threadIdx.x;
  if (idx >= (long long)M * N) return;
  int m = (int)(idx / N), n = (int)(idx % N);
  float acc = b ? b[n] : 0.f;
  for (int k = 0; k < K; ++k) acc += A[(size_t)m * K + k] * W[(size_t)k * N + n];
  if (act) acc = elu1(acc);
  C[idx] = acc;
}

// ---------------------------------------------------------------------------
// Elementwise / scatter kernels
// ---------------------------------------------------------------------------
__global__ void k_fill(float* __restrict__ p, float v, long long n) {
  long long i = (long long)blockIdx.x * blockDim.x + threadIdx.x;
  if (i < n) p[i] = v;
}

__global__ void k_deg(const int* __restrict__ dst, float* __restrict__ deg, int E) {
  int e = blockIdx.x * blockDim.x + threadIdx.x;
  if (e < E) atomicAdd(&deg[dst[e]], 1.f);
}

__global__ void k_scatter_rows(const float* __restrict__ ea,
                               const int* __restrict__ dst,
                               float* __restrict__ sum, long long total) {
  long long idx = (long long)blockIdx.x * blockDim.x + threadIdx.x;
  if (idx >= total) return;
  int e = (int)(idx / HID), c = (int)(idx % HID);
  atomicAdd(&sum[(size_t)dst[e] * HID + c], ea[idx]);
}

__global__ void k_div_deg(float* __restrict__ sum, const float* __restrict__ deg,
                          long long total) {
  long long idx = (long long)blockIdx.x * blockDim.x + threadIdx.x;
  if (idx >= total) return;
  int n = (int)(idx / HID);
  sum[idx] = sum[idx] / fmaxf(deg[n], 1.f);
}

// attention logits: per (edge, head) 32-dim leaky_relu dot with att vector
__global__ void k_logits(const float* __restrict__ xl, const float* __restrict__ xr,
                         const float* __restrict__ ef,
                         const int* __restrict__ src, const int* __restrict__ dst,
                         const float* __restrict__ att,
                         float* __restrict__ logit, int total) {
  int idx = blockIdx.x * blockDim.x + threadIdx.x;
  if (idx >= total) return;
  int e = idx / HEADS, hh = idx % HEADS;
  int s, d;
  if (e < N_EDGES) { s = src[e]; d = dst[e]; } else { s = d = e - N_EDGES; }
  const float* pl = xl + (size_t)s * HID + hh * DHEAD;
  const float* pr = xr + (size_t)d * HID + hh * DHEAD;
  const float* pe = ef + (size_t)e * HID + hh * DHEAD;
  const float* pa = att + hh * DHEAD;
  float acc = 0.f;
#pragma unroll 8
  for (int i = 0; i < DHEAD; ++i) {
    float m = pl[i] + pr[i] + pe[i];
    m = m > 0.f ? m : 0.2f * m;             // leaky_relu(0.2)
    acc += m * pa[i];
  }
  logit[idx] = acc;
}

__global__ void k_segmax(const float* __restrict__ logit,
                         const int* __restrict__ dst,
                         float* __restrict__ mx, int total) {
  int idx = blockIdx.x * blockDim.x + threadIdx.x;
  if (idx >= total) return;
  int e = idx / HEADS, hh = idx % HEADS;
  int d = (e < N_EDGES) ? dst[e] : e - N_EDGES;
  atomicMaxF(&mx[(size_t)d * HEADS + hh], logit[idx]);
}

__global__ void k_expden(float* __restrict__ logit, const int* __restrict__ dst,
                         const float* __restrict__ mx, float* __restrict__ den,
                         int total) {
  int idx = blockIdx.x * blockDim.x + threadIdx.x;
  if (idx >= total) return;
  int e = idx / HEADS, hh = idx % HEADS;
  int d = (e < N_EDGES) ? dst[e] : e - N_EDGES;
  float ex = __expf(logit[idx] - mx[(size_t)d * HEADS + hh]);
  logit[idx] = ex;                          // overwrite with numerator
  atomicAdd(&den[(size_t)d * HEADS + hh], ex);
}

__global__ void k_scatter_msg(const float* __restrict__ ex,
                              const float* __restrict__ den,
                              const float* __restrict__ xl,
                              const int* __restrict__ src,
                              const int* __restrict__ dst,
                              float* __restrict__ agg, long long total) {
  long long idx = (long long)blockIdx.x * blockDim.x + threadIdx.x;
  if (idx >= total) return;
  int e = (int)(idx / HID), c = (int)(idx % HID);
  int hh = c / DHEAD;
  int s, d;
  if (e < N_EDGES) { s = src[e]; d = dst[e]; } else { s = d = e - N_EDGES; }
  float alpha = ex[(size_t)e * HEADS + hh] /
                fmaxf(den[(size_t)d * HEADS + hh], 1e-16f);
  atomicAdd(&agg[(size_t)d * HID + c], alpha * xl[(size_t)s * HID + c]);
}

__global__ void k_combine(const float* __restrict__ agg,
                          const float* __restrict__ gbias,
                          const float* __restrict__ res,
                          float* __restrict__ hout, long long total) {
  long long idx = (long long)blockIdx.x * blockDim.x + threadIdx.x;
  if (idx >= total) return;
  int c = (int)(idx % HID);
  hout[idx] = elu1(agg[idx] + gbias[c] + res[idx]);
}

__global__ void k_cnt(const int* __restrict__ batch, float* __restrict__ cnt, int n) {
  int i = blockIdx.x * blockDim.x + threadIdx.x;
  if (i < n) atomicAdd(&cnt[batch[i]], 1.f);
}

__global__ void k_poolsum(const float* __restrict__ h, const int* __restrict__ batch,
                          float* __restrict__ gsum, long long total) {
  long long idx = (long long)blockIdx.x * blockDim.x + threadIdx.x;
  if (idx >= total) return;
  int n = (int)(idx / HID), c = (int)(idx % HID);
  atomicAdd(&gsum[(size_t)batch[n] * HID + c], h[idx]);
}

__global__ void k_pooldiv(float* __restrict__ gsum, const float* __restrict__ cnt,
                          long long total) {
  long long idx = (long long)blockIdx.x * blockDim.x + threadIdx.x;
  if (idx >= total) return;
  int g = (int)(idx / HID);
  gsum[idx] = gsum[idx] / fmaxf(cnt[g], 1.f);
}

__global__ void k_softmax(const float* __restrict__ in, float* __restrict__ out,
                          int cols) {
  __shared__ float red[256];
  int r = blockIdx.x;
  const float* p = in + (size_t)r * cols;
  float mx = -3.0e38f;
  for (int c = threadIdx.x; c < cols; c += 256) mx = fmaxf(mx, p[c]);
  red[threadIdx.x] = mx; __syncthreads();
  for (int s = 128; s > 0; s >>= 1) {
    if (threadIdx.x < s) red[threadIdx.x] = fmaxf(red[threadIdx.x], red[threadIdx.x + s]);
    __syncthreads();
  }
  mx = red[0]; __syncthreads();
  float sum = 0.f;
  for (int c = threadIdx.x; c < cols; c += 256) sum += __expf(p[c] - mx);
  red[threadIdx.x] = sum; __syncthreads();
  for (int s = 128; s > 0; s >>= 1) {
    if (threadIdx.x < s) red[threadIdx.x] += red[threadIdx.x + s];
    __syncthreads();
  }
  float inv = 1.f / red[0];
  for (int c = threadIdx.x; c < cols; c += 256)
    out[(size_t)r * cols + c] = __expf(p[c] - mx) * inv;
}

// ---------------------------------------------------------------------------
// Host launcher
// ---------------------------------------------------------------------------
extern "C" void kernel_launch(void* const* d_in, const int* in_sizes, int n_in,
                              void* d_out, int out_size, void* d_ws, size_t ws_size,
                              hipStream_t stream) {
  const float* x         = (const float*)d_in[0];
  const float* edge_attr = (const float*)d_in[1];
  const int*   edge_index= (const int*)d_in[2];
  const int*   batch     = (const int*)d_in[3];
  const float* node_W    = (const float*)d_in[4];
  const float* node_b    = (const float*)d_in[5];
  const float* edge_W    = (const float*)d_in[6];
  const float* edge_b    = (const float*)d_in[7];
  const float* Wl        = (const float*)d_in[8];
  const float* Wr        = (const float*)d_in[9];
  const float* We        = (const float*)d_in[10];
  const float* att       = (const float*)d_in[11];
  const float* gat_bias  = (const float*)d_in[12];
  const float* res_W     = (const float*)d_in[13];
  const float* res_b     = (const float*)d_in[14];
  const float* h1_W      = (const float*)d_in[15];
  const float* h1_b      = (const float*)d_in[16];
  const float* h2_W      = (const float*)d_in[17];
  const float* h2_b      = (const float*)d_in[18];
  const float* h3_W      = (const float*)d_in[19];
  const float* h3_b      = (const float*)d_in[20];

  const int* src = edge_index;
  const int* dst = edge_index + N_EDGES;

  // ---- workspace carve-out ----
  char* w = (char*)d_ws;
  auto alloc = [&](size_t nfloats) -> float* {
    float* p = (float*)w; w += nfloats * sizeof(float); return p;
  };
  float* hA     = alloc((size_t)N_NODES * HID);
  float* hB     = alloc((size_t)N_NODES * HID);
  float* ea     = alloc((size_t)N_EDGES * HID);
  float* loopat = alloc((size_t)N_NODES * HID);
  float* xl     = alloc((size_t)N_NODES * HID);
  float* xr     = alloc((size_t)N_NODES * HID);
  float* ef     = alloc((size_t)EA * HID);
  float* res    = alloc((size_t)N_NODES * HID);
  float* agg    = alloc((size_t)N_NODES * HID);
  float* deg    = alloc(N_NODES);
  float* logit  = alloc((size_t)EA * HEADS);
  float* mx     = alloc((size_t)N_NODES * HEADS);
  float* den    = alloc((size_t)N_NODES * HEADS);
  float* gsum   = alloc((size_t)N_GRAPHS * HID);
  float* cnt    = alloc(N_GRAPHS);
  float* g1     = alloc((size_t)N_GRAPHS * 2 * HID);
  float* g2     = alloc((size_t)N_GRAPHS * HID);
  float* glog   = alloc((size_t)N_GRAPHS * OUT);
  (void)ws_size; (void)in_sizes; (void)n_in; (void)out_size;

  auto blk = [](long long n) { return (unsigned)((n + 255) / 256); };
  auto fill = [&](float* p, float v, long long n) {
    k_fill<<<blk(n), 256, 0, stream>>>(p, v, n);
  };
  auto gemm = [&](const float* A, const float* B, const float* bias, float* C,
                  int M, int K, int N, int act) {
    dim3 g((M + 127) / 128, (N + 127) / 128);
    k_gemm_bf16<<<g, 256, 0, stream>>>(A, B, bias, C, M, K, N, act);
  };

  const long long NH  = (long long)N_NODES * HID;
  const long long EH  = (long long)N_EDGES * HID;
  const long long EAH = (long long)EA * HID;
  const long long EAh = (long long)EA * HEADS;

  // ---- encoders ----
  k_linear_small<<<blk(NH), 256, 0, stream>>>(x, node_W, node_b, hA,
                                              N_NODES, ND, HID, 1);
  k_linear_small<<<blk(EH), 256, 0, stream>>>(edge_attr, edge_W, edge_b, ea,
                                              N_EDGES, ED, HID, 1);

  // ---- self-loop attr = mean of encoded edge_attr per dst ----
  fill(deg, 0.f, N_NODES);
  fill(loopat, 0.f, NH);
  k_deg<<<blk(N_EDGES), 256, 0, stream>>>(dst, deg, N_EDGES);
  k_scatter_rows<<<blk(EH), 256, 0, stream>>>(ea, dst, loopat, EH);
  k_div_deg<<<blk(NH), 256, 0, stream>>>(loopat, deg, NH);

  // ---- GAT layers ----
  float* hcur = hA;
  float* hnxt = hB;
  for (int i = 0; i < LAYERS; ++i) {
    const size_t wo = (size_t)i * HID * HID;
    gemm(hcur,   Wl + wo, nullptr,           xl,  N_NODES, HID, HID, 0);
    gemm(hcur,   Wr + wo, nullptr,           xr,  N_NODES, HID, HID, 0);
    gemm(ea,     We + wo, nullptr,           ef,  N_EDGES, HID, HID, 0);
    gemm(loopat, We + wo, nullptr, ef + (size_t)N_EDGES * HID,
                                             N_NODES, HID, HID, 0);
    gemm(hcur, res_W + wo, res_b + (size_t)i * HID, res, N_NODES, HID, HID, 0);

    fill(mx, -3.0e38f, (long long)N_NODES * HEADS);
    fill(den, 0.f,     (long long)N_NODES * HEADS);
    fill(agg, 0.f, NH);

    k_logits<<<blk(EAh), 256, 0, stream>>>(xl, xr, ef, src, dst,
                                           att + (size_t)i * HEADS * DHEAD,
                                           logit, (int)EAh);
    k_segmax<<<blk(EAh), 256, 0, stream>>>(logit, dst, mx, (int)EAh);
    k_expden<<<blk(EAh), 256, 0, stream>>>(logit, dst, mx, den, (int)EAh);
    k_scatter_msg<<<blk(EAH), 256, 0, stream>>>(logit, den, xl, src, dst, agg, EAH);
    k_combine<<<blk(NH), 256, 0, stream>>>(agg, gat_bias + (size_t)i * HID,
                                           res, hnxt, NH);
    float* t = hcur; hcur = hnxt; hnxt = t;
  }

  // ---- mean pooling per graph ----
  fill(cnt, 0.f, N_GRAPHS);
  fill(gsum, 0.f, (long long)N_GRAPHS * HID);
  k_cnt<<<blk(N_NODES), 256, 0, stream>>>(batch, cnt, N_NODES);
  k_poolsum<<<blk(NH), 256, 0, stream>>>(hcur, batch, gsum, NH);
  k_pooldiv<<<blk((long long)N_GRAPHS * HID), 256, 0, stream>>>(
      gsum, cnt, (long long)N_GRAPHS * HID);

  // ---- MLP head + softmax ----
  gemm(gsum, h1_W, h1_b, g1, N_GRAPHS, HID, 2 * HID, 1);
  gemm(g1,   h2_W, h2_b, g2, N_GRAPHS, 2 * HID, HID, 1);
  gemm(g2,   h3_W, h3_b, glog, N_GRAPHS, HID, OUT, 0);
  k_softmax<<<N_GRAPHS, 256, 0, stream>>>(glog, (float*)d_out, OUT);
}
